// ActorCriticWithContext_16415365005729
// MI455X (gfx1250) — matrix operand order repaired and checked by reading, not compiled, not run on hardware
//
#include <hip/hip_runtime.h>
#include <hip/hip_bf16.h>
#include <cstdint>

// ---------------------------------------------------------------------------
// ActorCriticWithContext for MI455X (gfx1250), wave32 + WMMA bf16.
//   T=256 B=1024 H=W=9 C=3 CTX=64 HID=256 ACT=7
//   emb dim 853 -> padded to 864 (27 chunks of K=32)
// Flagship GEMM uses async global->LDS (ASYNCcnt) with double buffering.
// ---------------------------------------------------------------------------

#define T_DIM   256
#define B_DIM   1024
#define HID     256
#define NGATE   1024            // 4*HID
#define KPAD    864             // 853 padded to 27*32
#define M_TOT   (T_DIM * B_DIM) // 262144

typedef __attribute__((ext_vector_type(16))) __bf16 v16bf;
typedef __attribute__((ext_vector_type(8)))  __bf16 v8bf;
typedef __attribute__((ext_vector_type(8)))  float  v8f;
typedef int v4i __attribute__((vector_size(16)));   // matches builtin param type

#define AS1 __attribute__((address_space(1)))
#define AS3 __attribute__((address_space(3)))

#if __has_builtin(__builtin_amdgcn_global_load_async_to_lds_b128)
#define HAS_ASYNC_LDS 1
#else
#define HAS_ASYNC_LDS 0
#endif

__device__ __forceinline__ v8f wmma_bf16(v16bf a, v16bf b, v8f c) {
  // (neg_a, A, neg_b, B, c_mod, C, reuse_a, reuse_b)
  return __builtin_amdgcn_wmma_f32_16x16x32_bf16(false, a, false, b, (short)0, c,
                                                 false, false);
}

// Stage 16 bf16 (32B) global -> LDS. Async path: two b128 transfers handled by
// the async engine (no VGPR round-trip, tracked by ASYNCcnt).
__device__ __forceinline__ void stage16(const __bf16* g, __bf16* l) {
#if HAS_ASYNC_LDS
  __builtin_amdgcn_global_load_async_to_lds_b128((AS1 v4i*)g, (AS3 v4i*)l, 0, 0);
  __builtin_amdgcn_global_load_async_to_lds_b128((AS1 v4i*)(g + 8),
                                                 (AS3 v4i*)(l + 8), 0, 0);
#else
  *(v8bf*)l       = *(const v8bf*)g;
  *(v8bf*)(l + 8) = *(const v8bf*)(g + 8);
#endif
}

__device__ __forceinline__ void wait_async0() {
#if __has_builtin(__builtin_amdgcn_s_wait_asynccnt)
  __builtin_amdgcn_s_wait_asynccnt(0);
#elif HAS_ASYNC_LDS
  asm volatile("s_wait_asynccnt 0x0" ::: "memory");
#endif
}

// A fragment (16x32 bf16): lane m = l&15; K = kb..kb+15 contiguous with
// kb = (l>>4)*16. Tile is row-major, ldk elems.
__device__ __forceinline__ v16bf load_a_frag(const __bf16* tile, int row0, int ldk) {
  const int lane = threadIdx.x & 31;
  const int m  = lane & 15;
  const int kb = (lane >> 4) << 4;
  const __bf16* p = tile + (row0 + m) * ldk + kb;
  v8bf lo = *(const v8bf*)p;
  v8bf hi = *(const v8bf*)(p + 8);
  return __builtin_shufflevector(lo, hi, 0,1,2,3,4,5,6,7,8,9,10,11,12,13,14,15);
}

// B fragment (32x16 bf16): lane n = l&15; K = kb..kb+15 contiguous with
// kb = (l>>4)*16. Tile is stored transposed ([n][k], K contiguous).
__device__ __forceinline__ v16bf load_b_frag(const __bf16* tileT, int col0, int ldk,
                                             int koff) {
  const int lane = threadIdx.x & 31;
  const int n  = lane & 15;
  const int kb = (lane >> 4) << 4;
  return *(const v16bf*)(tileT + (col0 + n) * ldk + koff + kb);
}

__device__ __forceinline__ float sigm(float x) { return 1.0f / (1.0f + __expf(-x)); }

// ---------------------------------------------------------------------------
// Weight preparation (f32 -> bf16, transposed so K is contiguous per column)
// ---------------------------------------------------------------------------
__global__ __launch_bounds__(256) void cvt_wi_kernel(const float* __restrict__ wi,
                                                     __bf16* __restrict__ wiT) {
  int idx = blockIdx.x * 256 + threadIdx.x;      // over NGATE*KPAD
  if (idx >= NGATE * KPAD) return;
  int n = idx / KPAD, k = idx % KPAD;
  float v = (k < 853) ? wi[(size_t)k * NGATE + n] : 0.0f;
  wiT[idx] = (__bf16)v;
}

__global__ __launch_bounds__(256) void cvt_wh_kernel(const float* __restrict__ wh,
                                                     __bf16* __restrict__ whT) {
  int idx = blockIdx.x * 256 + threadIdx.x;      // over NGATE*HID
  int n = idx / HID, k = idx % HID;
  whT[idx] = (__bf16)wh[(size_t)k * NGATE + n];
}

__global__ __launch_bounds__(256) void cvt_heads_kernel(const float* __restrict__ a0_k,
                                                        const float* __restrict__ cr0_k,
                                                        __bf16* __restrict__ w0T) {
  int idx = blockIdx.x * 256 + threadIdx.x;      // over 64*HID
  int j = idx / HID, k = idx % HID;
  float v = (j < 32) ? a0_k[(size_t)k * 32 + j] : cr0_k[(size_t)k * 32 + (j - 32)];
  w0T[idx] = (__bf16)v;
}

__global__ __launch_bounds__(256) void init_state_kernel(const float* __restrict__ c0,
                                                         const float* __restrict__ h0,
                                                         float* __restrict__ c_ws,
                                                         __bf16* __restrict__ h_bf) {
  int idx = blockIdx.x * 256 + threadIdx.x;      // over B_DIM*HID
  c_ws[idx] = c0[idx];
  h_bf[idx] = (__bf16)h0[idx];
}

// ---------------------------------------------------------------------------
// Conv(3x3x3->16, VALID) + ReLU + dir embedding + context concat -> emb bf16
// One block per (t,b) sample; emb row stride KPAD with zero padding.
// ---------------------------------------------------------------------------
__global__ __launch_bounds__(256) void conv_embed_kernel(
    const float* __restrict__ image, const int* __restrict__ agent_dir,
    const float* __restrict__ context, const float* __restrict__ conv_k,
    const float* __restrict__ conv_b, const float* __restrict__ emb_k,
    const float* __restrict__ emb_b, __bf16* __restrict__ emb) {
  const int m = blockIdx.x;                      // 0..M_TOT-1, m = t*B + b
  const int tid = threadIdx.x;
  __shared__ float simg[243];                    // 9*9*3
  __shared__ float sker[432];                    // 3*3*3*16
  for (int i = tid; i < 243; i += 256) simg[i] = image[(size_t)m * 243 + i];
  for (int i = tid; i < 432; i += 256) sker[i] = conv_k[i];
  __syncthreads();

  __bf16* row = emb + (size_t)m * KPAD;
  for (int k = tid; k < 784; k += 256) {         // conv outputs (oy,ox,oc)
    const int oc = k & 15;
    const int os = k >> 4;
    const int ox = os % 7;
    const int oy = os / 7;
    float acc = conv_b[oc];
    #pragma unroll
    for (int ky = 0; ky < 3; ++ky)
      #pragma unroll
      for (int kx = 0; kx < 3; ++kx)
        #pragma unroll
        for (int ic = 0; ic < 3; ++ic)
          acc += simg[((oy + ky) * 9 + (ox + kx)) * 3 + ic] *
                 sker[((ky * 3 + kx) * 3 + ic) * 16 + oc];
    row[k] = (__bf16)fmaxf(acc, 0.0f);
  }
  if (tid < 80) {                                // dir(5) + ctx(64) + pad(11)
    float v;
    if (tid < 5)       v = emb_k[agent_dir[m] * 5 + tid] + emb_b[tid];
    else if (tid < 69) v = context[(size_t)(m & (B_DIM - 1)) * 64 + (tid - 5)];
    else               v = 0.0f;
    row[784 + tid] = (__bf16)v;
  }
}

// ---------------------------------------------------------------------------
// GEMM 1: xg[M_TOT x 1024] = emb[M_TOT x 864] @ wi[864 x 1024] + b_lstm
// Block tile 128x128, 8 waves, each wave 32x64 (2x4 WMMA tiles), K step 32.
// Async global->LDS double buffering: fetch of tile k+1 overlaps WMMA on k.
// ---------------------------------------------------------------------------
__global__ __launch_bounds__(256) void gemm_xg_kernel(
    const __bf16* __restrict__ emb, const __bf16* __restrict__ wiT,
    const float* __restrict__ b_lstm, float* __restrict__ xg) {
  __shared__ __align__(16) __bf16 sA[2][128 * 32];
  __shared__ __align__(16) __bf16 sB[2][128 * 32];
  const int tid = threadIdx.x, lane = tid & 31, wave = tid >> 5;
  const int wm = wave & 3, wn = wave >> 2;
  const int m0 = blockIdx.x * 128, n0 = blockIdx.y * 128;
  const int r = tid >> 1, half = tid & 1;
  const int dst = r * 32 + half * 16;
  const __bf16* ga = emb + (size_t)(m0 + r) * KPAD + half * 16;
  const __bf16* gb = wiT + (size_t)(n0 + r) * KPAD + half * 16;
  v8f acc[2][4] = {};

  // prologue: fetch K-tile 0 into buffer 0
  stage16(ga, &sA[0][dst]);
  stage16(gb, &sB[0][dst]);

  int cur = 0;
  for (int kc = 0; kc < KPAD; kc += 32) {
    wait_async0();               // own async transfers landed in LDS
    __syncthreads();             // all waves' tile data visible
    if (kc + 32 < KPAD) {        // kick off next K-tile into other buffer
      stage16(ga + kc + 32, &sA[cur ^ 1][dst]);
      stage16(gb + kc + 32, &sB[cur ^ 1][dst]);
    }
    v16bf a0 = load_a_frag(sA[cur], wm * 32, 32);
    v16bf a1 = load_a_frag(sA[cur], wm * 32 + 16, 32);
    #pragma unroll
    for (int ni = 0; ni < 4; ++ni) {
      v16bf b = load_b_frag(sB[cur], wn * 64 + ni * 16, 32, 0);
      acc[0][ni] = wmma_bf16(a0, b, acc[0][ni]);
      acc[1][ni] = wmma_bf16(a1, b, acc[1][ni]);
    }
    __syncthreads();             // reads done before this buffer is refilled
    cur ^= 1;
  }
  const int mlo = (lane >> 4) << 3, nn = lane & 15;
  #pragma unroll
  for (int mi = 0; mi < 2; ++mi)
    #pragma unroll
    for (int ni = 0; ni < 4; ++ni) {
      const int gn = n0 + wn * 64 + ni * 16 + nn;
      const float bias = b_lstm[gn];
      #pragma unroll
      for (int rr = 0; rr < 8; ++rr) {
        const int gm = m0 + wm * 32 + mi * 16 + rr + mlo;
        xg[(size_t)gm * NGATE + gn] = acc[mi][ni][rr] + bias;
      }
    }
}

// ---------------------------------------------------------------------------
// GEMM 2 (per step t): gates[1024 x 1024] = xg_t + h_masked @ wh
// h masking (done -> h0) fused into the A-tile stage.
// ---------------------------------------------------------------------------
__global__ __launch_bounds__(256) void gemm_lstm_kernel(
    const float* __restrict__ xg_t, const __bf16* __restrict__ h_bf,
    const float* __restrict__ h0, const unsigned char* __restrict__ dones_t,
    const __bf16* __restrict__ whT, float* __restrict__ gates) {
  __shared__ __align__(16) __bf16 sA[128 * 32];
  __shared__ __align__(16) __bf16 sB[128 * 32];
  const int tid = threadIdx.x, lane = tid & 31, wave = tid >> 5;
  const int wm = wave & 3, wn = wave >> 2;
  const int m0 = blockIdx.x * 128, n0 = blockIdx.y * 128;
  const int r = tid >> 1, half = tid & 1;
  const int bidx = m0 + r;
  const bool dn = dones_t[bidx] != 0;
  v8f acc[2][4] = {};

  for (int kc = 0; kc < HID; kc += 32) {
    if (dn) {
      const float* src = h0 + (size_t)bidx * HID + kc + half * 16;
      #pragma unroll
      for (int i = 0; i < 16; ++i)
        sA[r * 32 + half * 16 + i] = (__bf16)src[i];
    } else {
      const __bf16* src = h_bf + (size_t)bidx * HID + kc + half * 16;
      *(v8bf*)(sA + r * 32 + half * 16)     = *(const v8bf*)src;
      *(v8bf*)(sA + r * 32 + half * 16 + 8) = *(const v8bf*)(src + 8);
    }
    const __bf16* gb = whT + (size_t)(n0 + r) * HID + kc + half * 16;
    *(v8bf*)(sB + r * 32 + half * 16)     = *(const v8bf*)gb;
    *(v8bf*)(sB + r * 32 + half * 16 + 8) = *(const v8bf*)(gb + 8);
    __syncthreads();
    v16bf a0 = load_a_frag(sA, wm * 32, 32);
    v16bf a1 = load_a_frag(sA, wm * 32 + 16, 32);
    #pragma unroll
    for (int ni = 0; ni < 4; ++ni) {
      v16bf b = load_b_frag(sB, wn * 64 + ni * 16, 32, 0);
      acc[0][ni] = wmma_bf16(a0, b, acc[0][ni]);
      acc[1][ni] = wmma_bf16(a1, b, acc[1][ni]);
    }
    __syncthreads();
  }
  const int mlo = (lane >> 4) << 3, nn = lane & 15;
  #pragma unroll
  for (int mi = 0; mi < 2; ++mi)
    #pragma unroll
    for (int ni = 0; ni < 4; ++ni)
      #pragma unroll
      for (int rr = 0; rr < 8; ++rr) {
        const int gm = m0 + wm * 32 + mi * 16 + rr + mlo;
        const int gn = n0 + wn * 64 + ni * 16 + nn;
        gates[(size_t)gm * NGATE + gn] =
            acc[mi][ni][rr] + xg_t[(size_t)gm * NGATE + gn];
      }
}

// ---------------------------------------------------------------------------
// LSTM gate elementwise (per step t): c update, ys[t] = new_h, h_bf = bf16(new_h)
// ---------------------------------------------------------------------------
__global__ __launch_bounds__(256) void lstm_gates_kernel(
    const float* __restrict__ gates, const float* __restrict__ c0,
    const unsigned char* __restrict__ dones_t, float* __restrict__ c_ws,
    float* __restrict__ ys_t, __bf16* __restrict__ h_bf) {
  const int idx = blockIdx.x * 256 + threadIdx.x;  // over B_DIM*HID
  const int b = idx >> 8, j = idx & (HID - 1);
  const float gi = gates[(size_t)b * NGATE + j];
  const float gf = gates[(size_t)b * NGATE + HID + j];
  const float gg = gates[(size_t)b * NGATE + 2 * HID + j];
  const float go = gates[(size_t)b * NGATE + 3 * HID + j];
  const float c  = dones_t[b] ? c0[idx] : c_ws[idx];
  const float nc = sigm(gf) * c + sigm(gi) * tanhf(gg);
  const float nh = sigm(go) * tanhf(nc);
  c_ws[idx] = nc;
  ys_t[idx] = nh;
  h_bf[idx] = (__bf16)nh;
}

// ---------------------------------------------------------------------------
// Heads: hid = relu(ys @ [a0_k|cr0_k] + b) via WMMA (M=128/block, N=64, K=256),
// then fused tiny second layers -> logits (7) and value (1).
// ---------------------------------------------------------------------------
__global__ __launch_bounds__(256) void heads_kernel(
    const float* __restrict__ ys, const __bf16* __restrict__ w0T,
    const float* __restrict__ a0_b, const float* __restrict__ cr0_b,
    const float* __restrict__ a1_k, const float* __restrict__ a1_b,
    const float* __restrict__ cr1_k, const float* __restrict__ cr1_b,
    float* __restrict__ out_logits, float* __restrict__ out_value) {
  __shared__ __align__(16) char smem[8192 + 32768];
  __bf16* sA = (__bf16*)smem;                    // 128 x 32 bf16   (8 KB)
  __bf16* sB = (__bf16*)(smem + 8192);           // 64 x 256 bf16   (32 KB)
  float*  sH = (float*)smem;                     // 128 x 64 f32, reused (32 KB)

  const int tid = threadIdx.x, lane = tid & 31, wave = tid >> 5;
  const int m0 = blockIdx.x * 128;
  const int r = tid >> 1, half = tid & 1;

  for (int i = tid; i < 64 * HID; i += 256) sB[i] = w0T[i];  // whole B resident

  v8f acc[4] = {};
  for (int kc = 0; kc < HID; kc += 32) {
    const float* src = ys + (size_t)(m0 + r) * HID + kc + half * 16;
    #pragma unroll
    for (int i = 0; i < 16; ++i)
      sA[r * 32 + half * 16 + i] = (__bf16)src[i];
    __syncthreads();
    v16bf a = load_a_frag(sA, wave * 16, 32);
    #pragma unroll
    for (int ni = 0; ni < 4; ++ni) {
      v16bf b = load_b_frag(sB, ni * 16, HID, kc);
      acc[ni] = wmma_bf16(a, b, acc[ni]);
    }
    __syncthreads();
  }

  // relu + bias -> shared hidden (aliases sA/sB; all WMMA reads are done)
  const int mlo = (lane >> 4) << 3, nn = lane & 15;
  #pragma unroll
  for (int ni = 0; ni < 4; ++ni) {
    const int hn = ni * 16 + nn;
    const float bias = (hn < 32) ? a0_b[hn] : cr0_b[hn - 32];
    #pragma unroll
    for (int rr = 0; rr < 8; ++rr) {
      const int hm = wave * 16 + rr + mlo;
      sH[hm * 64 + hn] = fmaxf(acc[ni][rr] + bias, 0.0f);
    }
  }
  __syncthreads();

  // second layers: 128 rows x (7 logits + 1 value), 32-length dots
  for (int task = tid; task < 128 * 8; task += 256) {
    const int row = task >> 3, out = task & 7;
    const float* h = sH + row * 64;
    if (out < 7) {
      float s = a1_b[out];
      #pragma unroll
      for (int q = 0; q < 32; ++q) s += h[q] * a1_k[q * 7 + out];
      out_logits[(size_t)(m0 + row) * 7 + out] = s;
    } else {
      float s = cr1_b[0];
      #pragma unroll
      for (int q = 0; q < 32; ++q) s += h[32 + q] * cr1_k[q];
      out_value[m0 + row] = s;
    }
  }
}

// ---------------------------------------------------------------------------
// Launch
// ---------------------------------------------------------------------------
extern "C" void kernel_launch(void* const* d_in, const int* in_sizes, int n_in,
                              void* d_out, int out_size, void* d_ws, size_t ws_size,
                              hipStream_t stream) {
  const float* image     = (const float*)d_in[0];
  const int*   agent_dir = (const int*)d_in[1];
  const unsigned char* dones = (const unsigned char*)d_in[2];  // jnp bool (1B)
  const float* context   = (const float*)d_in[3];
  const float* c0        = (const float*)d_in[4];
  const float* h0        = (const float*)d_in[5];
  const float* conv_k    = (const float*)d_in[6];
  const float* conv_b    = (const float*)d_in[7];
  const float* emb_k     = (const float*)d_in[8];
  const float* emb_b     = (const float*)d_in[9];
  const float* wi        = (const float*)d_in[10];
  const float* wh        = (const float*)d_in[11];
  const float* b_lstm    = (const float*)d_in[12];
  const float* a0_k      = (const float*)d_in[13];
  const float* a0_b      = (const float*)d_in[14];
  const float* a1_k      = (const float*)d_in[15];
  const float* a1_b      = (const float*)d_in[16];
  const float* cr0_k     = (const float*)d_in[17];
  const float* cr0_b     = (const float*)d_in[18];
  const float* cr1_k     = (const float*)d_in[19];
  const float* cr1_b     = (const float*)d_in[20];

  uint8_t* ws = (uint8_t*)d_ws;
  size_t off = 0;
  auto take = [&](size_t bytes) -> void* {
    void* p = ws + off;
    off += (bytes + 255) & ~(size_t)255;
    return p;
  };
  __bf16* wiT   = (__bf16*)take((size_t)KPAD * NGATE * 2);
  __bf16* whT   = (__bf16*)take((size_t)HID * NGATE * 2);
  __bf16* w0T   = (__bf16*)take((size_t)64 * HID * 2);
  __bf16* embB  = (__bf16*)take((size_t)M_TOT * KPAD * 2);
  float*  xg    = (float*) take((size_t)M_TOT * NGATE * 4);
  float*  ys    = (float*) take((size_t)M_TOT * HID * 4);
  float*  gates = (float*) take((size_t)B_DIM * NGATE * 4);
  float*  c_ws  = (float*) take((size_t)B_DIM * HID * 4);
  __bf16* h_bf  = (__bf16*)take((size_t)B_DIM * HID * 2);

  // weight prep + state init
  cvt_wi_kernel<<<(KPAD * NGATE + 255) / 256, 256, 0, stream>>>(wi, wiT);
  cvt_wh_kernel<<<(HID * NGATE) / 256, 256, 0, stream>>>(wh, whT);
  cvt_heads_kernel<<<(64 * HID) / 256, 256, 0, stream>>>(a0_k, cr0_k, w0T);
  init_state_kernel<<<(B_DIM * HID) / 256, 256, 0, stream>>>(c0, h0, c_ws, h_bf);

  // conv + embed, then the big input GEMM
  conv_embed_kernel<<<M_TOT, 256, 0, stream>>>(image, agent_dir, context, conv_k,
                                               conv_b, emb_k, emb_b, embB);
  gemm_xg_kernel<<<dim3(M_TOT / 128, NGATE / 128), 256, 0, stream>>>(embB, wiT,
                                                                     b_lstm, xg);
  // LSTM scan: 256 dependent (GEMM, gate) pairs on the stream
  for (int t = 0; t < T_DIM; ++t) {
    const float* xg_t = xg + (size_t)t * B_DIM * NGATE;
    const unsigned char* dn_t = dones + (size_t)t * B_DIM;
    gemm_lstm_kernel<<<dim3(B_DIM / 128, NGATE / 128), 256, 0, stream>>>(
        xg_t, h_bf, h0, dn_t, whT, gates);
    lstm_gates_kernel<<<(B_DIM * HID) / 256, 256, 0, stream>>>(
        gates, c0, dn_t, c_ws, ys + (size_t)t * B_DIM * HID, h_bf);
  }

  // heads (logits at +524288 floats, value at +2359296 floats)
  float* out = (float*)d_out;
  heads_kernel<<<M_TOT / 128, 256, 0, stream>>>(ys, w0T, a0_b, cr0_b, a1_k, a1_b,
                                                cr1_k, cr1_b, out + 524288,
                                                out + 2359296);

  // c_fin, h_fin
  (void)hipMemcpyAsync(out, c_ws, (size_t)B_DIM * HID * 4,
                       hipMemcpyDeviceToDevice, stream);
  (void)hipMemcpyAsync(out + B_DIM * HID, ys + (size_t)255 * B_DIM * HID,
                       (size_t)B_DIM * HID * 4, hipMemcpyDeviceToDevice, stream);
}
